// Head_47871705481337
// MI455X (gfx1250) — matrix operand order repaired and checked by reading, not compiled, not run on hardware
//
#include <hip/hip_runtime.h>
#include <hip/hip_bf16.h>

// ---------------------------------------------------------------------------
// Single-head causal attention, fused flash-attention style, for gfx1250.
//   x:[8,2048,1024] f32, Wq/Wk/Wv:[1024,64] f32  ->  out:[8,2048,64] f32
// All matmuls run on v_wmma_f32_16x16x32_bf16 (f32 accumulate).
// ---------------------------------------------------------------------------

#define B_  8
#define T_  2048
#define C_  1024
#define H_  64

typedef __attribute__((ext_vector_type(16))) __bf16 v16bf;
typedef __attribute__((ext_vector_type(8)))  __bf16 v8bf;
typedef __attribute__((ext_vector_type(8)))  float  v8f;

// sched_group_barrier masks
#define SG_WMMA 0x008
#define SG_VMEM 0x010

// --------------------------- phase 0: W -> Wt(bf16, [H][C]) -----------------
__global__ void wt_kernel(const float* __restrict__ W, __bf16* __restrict__ Wt) {
  int idx = blockIdx.x * blockDim.x + threadIdx.x;   // 65536 threads
  int h = idx & (H_ - 1);
  int c = idx >> 6;
  Wt[h * C_ + c] = (__bf16)W[c * H_ + h];
}

// --------------------------- phase 1: q/k/v projections ---------------------
// One wave per 16-row tile of x. x tile loaded once, reused for Wq/Wk/Wv.
// sched_group_barrier forces: all VMEM loads issue first, then the 12 WMMAs,
// so the waitcnt pass emits partial waits and loads overlap the XDL stream.
__global__ __launch_bounds__(32)
void proj_kernel(const float* __restrict__ x,
                 const __bf16* __restrict__ Wqt,
                 const __bf16* __restrict__ Wkt,
                 const __bf16* __restrict__ Wvt,
                 __bf16* __restrict__ qo,    // [B,T,H] bf16, *0.125*log2e
                 __bf16* __restrict__ ko,    // [B,T,H] bf16
                 __bf16* __restrict__ vto) { // [B,H,T] bf16 (transposed)
  const int lane = threadIdx.x;
  const int half = lane >> 4;
  const int lr   = lane & 15;
  const int b    = blockIdx.x >> 7;        // 128 row tiles per batch
  const int ti   = blockIdx.x & 127;
  const int row0 = ti * 16;

  const float* xrow = x + ((size_t)(b * T_ + row0 + lr)) * C_;

  v8f accQ[4], accK[4], accV[4];
#pragma unroll
  for (int n = 0; n < 4; n++)
#pragma unroll
    for (int j = 0; j < 8; j++) { accQ[n][j] = 0.f; accK[n][j] = 0.f; accV[n][j] = 0.f; }

  for (int kb = 0; kb < C_; kb += 32) {
    // prefetch next x chunk for this lane's row (global_prefetch_b8, free regs)
    if (kb + 32 < C_) __builtin_prefetch(xrow + kb + 32 + half * 8, 0, 1);

    // A fragment: 16x32 bf16, ISA layout (lanes 0-15: K=kb+0..7,16..23; 16-31: +8).
    v16bf a;
    const float* ap = xrow + kb + half * 8;
#pragma unroll
    for (int e = 0; e < 8; e++) a[e]     = (__bf16)ap[e];
#pragma unroll
    for (int e = 0; e < 8; e++) a[8 + e] = (__bf16)ap[16 + e];

    // Load all 12 B fragments (must stay in distinct registers).
    v16bf bq[4], bk[4], bv[4];
#pragma unroll
    for (int n = 0; n < 4; n++) {
      const size_t off = (size_t)(n * 16 + lr) * C_ + kb + half * 8;
      ((v8bf*)&bq[n])[0] = ((const v8bf*)(Wqt + off))[0];
      ((v8bf*)&bq[n])[1] = ((const v8bf*)(Wqt + off))[2];
      ((v8bf*)&bk[n])[0] = ((const v8bf*)(Wkt + off))[0];
      ((v8bf*)&bk[n])[1] = ((const v8bf*)(Wkt + off))[2];
      ((v8bf*)&bv[n])[0] = ((const v8bf*)(Wvt + off))[0];
      ((v8bf*)&bv[n])[1] = ((const v8bf*)(Wvt + off))[2];
    }
#pragma unroll
    for (int n = 0; n < 4; n++) {
      accQ[n] = __builtin_amdgcn_wmma_f32_16x16x32_bf16(false, a, false, bq[n], (short)0, accQ[n], false, false);
      accK[n] = __builtin_amdgcn_wmma_f32_16x16x32_bf16(false, a, false, bk[n], (short)0, accK[n], false, false);
      accV[n] = __builtin_amdgcn_wmma_f32_16x16x32_bf16(false, a, false, bv[n], (short)0, accV[n], false, false);
    }
    // Pipeline shape: all VMEM first (A + 24 B-frag loads + prefetch), then
    // the 12 WMMAs; conversion VALU floats in between by dependency.
    __builtin_amdgcn_sched_group_barrier(SG_VMEM, 32, 0);
    __builtin_amdgcn_sched_group_barrier(SG_WMMA, 12, 0);
  }

  // D layout: VGPR j -> row j + 8*half, column = lr (+16*n over H).
  // Fold 1/sqrt(H) and log2(e) into q so attention softmax can use v_exp_f32
  // (exp2) directly: softmax is invariant under the uniform base change.
  const float qscale = 0.125f * 1.44269504088896340736f;
#pragma unroll
  for (int n = 0; n < 4; n++) {
#pragma unroll
    for (int j = 0; j < 8; j++) {
      const int row = row0 + j + half * 8;
      const int h   = n * 16 + lr;
      const size_t qi = (size_t)(b * T_ + row) * H_ + h;
      qo[qi] = (__bf16)(accQ[n][j] * qscale);
      ko[qi] = (__bf16)accK[n][j];
      vto[(size_t)(b * H_ + h) * T_ + row] = (__bf16)accV[n][j];
    }
  }
}

// --------------------------- phase 2: flash attention -----------------------
// One wave per 16-row tile; 32 key columns per step. k/v fragment loads are
// hoisted ahead of the VALU softmax so they overlap; next step is prefetched.
__global__ __launch_bounds__(32)
void attn_kernel(const __bf16* __restrict__ qg,   // [B,T,H] bf16 (scaled)
                 const __bf16* __restrict__ kg,   // [B,T,H] bf16
                 const __bf16* __restrict__ vtg,  // [B,H,T] bf16
                 float* __restrict__ out) {       // [B,T,H] f32
  __shared__ __align__(16) __bf16 lds_p[16 * 32]; // P-tile transpose bounce

  const int lane = threadIdx.x;
  const int half = lane >> 4;
  const int lr   = lane & 15;
  const int b    = blockIdx.x >> 7;
  const int ti   = blockIdx.x & 127;
  const int row0 = ti * 16;

  // Q as two A fragments (h = 0..31, 32..63)
  v16bf qa[2];
  {
    const __bf16* qrow = qg + (size_t)(b * T_ + row0 + lr) * H_;
#pragma unroll
    for (int f = 0; f < 2; f++) {
      ((v8bf*)&qa[f])[0] = *(const v8bf*)(qrow + f * 32 + half * 8);
      ((v8bf*)&qa[f])[1] = *(const v8bf*)(qrow + f * 32 + 16 + half * 8);
    }
  }

  float m_run[8], l_run[8];
#pragma unroll
  for (int j = 0; j < 8; j++) { m_run[j] = -3.0e38f; l_run[j] = 0.f; }
  v8f accO[4];
#pragma unroll
  for (int n = 0; n < 4; n++)
#pragma unroll
    for (int j = 0; j < 8; j++) accO[n][j] = 0.f;

  const int jend = row0 + 16;                 // exclusive causal column bound
  for (int jj = 0; jj < jend; jj += 32) {
    // ---- load all k fragments for both 16-col tiles (distinct regs) ----
    v16bf kbf[2][2];
#pragma unroll
    for (int st = 0; st < 2; st++) {
      const __bf16* krow = kg + (size_t)(b * T_ + jj + st * 16 + lr) * H_;
#pragma unroll
      for (int f = 0; f < 2; f++) {
        ((v8bf*)&kbf[st][f])[0] = *(const v8bf*)(krow + f * 32 + half * 8);
        ((v8bf*)&kbf[st][f])[1] = *(const v8bf*)(krow + f * 32 + 16 + half * 8);
      }
    }
    // ---- issue v fragments now; they overlap S WMMAs + softmax VALU ----
    v16bf vbf[4];
#pragma unroll
    for (int n = 0; n < 4; n++) {
      const __bf16* vrow = vtg + (size_t)(b * H_ + n * 16 + lr) * T_ + jj + half * 8;
      ((v8bf*)&vbf[n])[0] = ((const v8bf*)vrow)[0];
      ((v8bf*)&vbf[n])[1] = ((const v8bf*)vrow)[2];
    }
    // ---- prefetch next step's k rows and v chunks ----
    if (jj + 32 < jend) {
      __builtin_prefetch(kg + (size_t)(b * T_ + jj + 32 + lr) * H_, 0, 1);
      __builtin_prefetch(kg + (size_t)(b * T_ + jj + 48 + lr) * H_, 0, 1);
#pragma unroll
      for (int n = 0; n < 4; n++)
        __builtin_prefetch(vtg + (size_t)(b * H_ + n * 16 + lr) * T_ + jj + 32 + half * 16, 0, 1);
    }

    // ---- S = q . k^T for two 16-col tiles (K-dim = 64 over h) ----
    v8f S[2];
#pragma unroll
    for (int st = 0; st < 2; st++) {
      v8f s;
#pragma unroll
      for (int j = 0; j < 8; j++) s[j] = 0.f;
#pragma unroll
      for (int f = 0; f < 2; f++)
        s = __builtin_amdgcn_wmma_f32_16x16x32_bf16(false, qa[f], false, kbf[st][f], (short)0, s, false, false);
      // causal mask (lane-local select; EXEC stays all-ones)
#pragma unroll
      for (int j = 0; j < 8; j++) {
        const int row = row0 + j + half * 8;
        const int col = jj + st * 16 + lr;
        s[j] = (col > row) ? -3.0e38f : s[j];
      }
      S[st] = s;
    }

    // ---- online softmax in base 2 (log2e folded into q scale) ----
#pragma unroll
    for (int j = 0; j < 8; j++) {
      float mx = fmaxf(S[0][j], S[1][j]);
#pragma unroll
      for (int d = 1; d < 16; d <<= 1) mx = fmaxf(mx, __shfl_xor(mx, d, 32));
      const float mnew = fmaxf(m_run[j], mx);
      const float sc   = exp2f(m_run[j] - mnew);
      m_run[j] = mnew;
      const float p0 = exp2f(S[0][j] - mnew);
      const float p1 = exp2f(S[1][j] - mnew);
      S[0][j] = p0; S[1][j] = p1;
      float rs = p0 + p1;
#pragma unroll
      for (int d = 1; d < 16; d <<= 1) rs += __shfl_xor(rs, d, 32);
      l_run[j] = l_run[j] * sc + rs;
#pragma unroll
      for (int n = 0; n < 4; n++) accO[n][j] *= sc;
    }

    // ---- transpose P (D layout -> A layout) through LDS ----
#pragma unroll
    for (int j = 0; j < 8; j++) {
      const int row = j + half * 8;
      lds_p[row * 32 + lr]      = (__bf16)S[0][j];
      lds_p[row * 32 + 16 + lr] = (__bf16)S[1][j];
    }
    // Pipeline shape for this scheduling region: all VMEM loads first, then
    // the 4 S WMMAs; softmax VALU and DS writes order by dependency.
    __builtin_amdgcn_sched_group_barrier(SG_VMEM, 24, 0);
    __builtin_amdgcn_sched_group_barrier(SG_WMMA, 4, 0);

    asm volatile("s_wait_dscnt 0" ::: "memory");   // CDNA5 split DS counter
    v16bf pa;
    ((v8bf*)&pa)[0] = *(const v8bf*)&lds_p[lr * 32 + half * 8];
    ((v8bf*)&pa)[1] = *(const v8bf*)&lds_p[lr * 32 + 16 + half * 8];

    // ---- O += P . V  (K=32 over the 32 key columns) ----
#pragma unroll
    for (int n = 0; n < 4; n++)
      accO[n] = __builtin_amdgcn_wmma_f32_16x16x32_bf16(false, pa, false, vbf[n], (short)0, accO[n], false, false);
  }

  // ---- normalize and write out ----
#pragma unroll
  for (int j = 0; j < 8; j++) {
    const float inv = 1.0f / l_run[j];
    const int row = row0 + j + half * 8;
#pragma unroll
    for (int n = 0; n < 4; n++)
      out[(size_t)(b * T_ + row) * H_ + n * 16 + lr] = accO[n][j] * inv;
  }
}

// ---------------------------------------------------------------------------
extern "C" void kernel_launch(void* const* d_in, const int* in_sizes, int n_in,
                              void* d_out, int out_size, void* d_ws, size_t ws_size,
                              hipStream_t stream) {
  const float* x  = (const float*)d_in[0];
  const float* Wq = (const float*)d_in[1];
  const float* Wk = (const float*)d_in[2];
  const float* Wv = (const float*)d_in[3];
  float* out = (float*)d_out;

  char* ws = (char*)d_ws;
  const size_t qkv_bytes = (size_t)B_ * T_ * H_ * sizeof(__bf16);   // 2 MB each
  __bf16* qo  = (__bf16*)(ws);
  __bf16* ko  = (__bf16*)(ws + qkv_bytes);
  __bf16* vto = (__bf16*)(ws + 2 * qkv_bytes);
  __bf16* wqt = (__bf16*)(ws + 3 * qkv_bytes);
  __bf16* wkt = wqt + (size_t)H_ * C_;
  __bf16* wvt = wkt + (size_t)H_ * C_;

  wt_kernel<<<(C_ * H_) / 256, 256, 0, stream>>>(Wq, wqt);
  wt_kernel<<<(C_ * H_) / 256, 256, 0, stream>>>(Wk, wkt);
  wt_kernel<<<(C_ * H_) / 256, 256, 0, stream>>>(Wv, wvt);
  proj_kernel<<<B_ * (T_ / 16), 32, 0, stream>>>(x, wqt, wkt, wvt, qo, ko, vto);
  attn_kernel<<<B_ * (T_ / 16), 32, 0, stream>>>(qo, ko, vto, out);
}
